// BinaryLinear_26714696581575
// MI455X (gfx1250) — matrix operand order repaired and checked by reading, not compiled, not run on hardware
//
#include <hip/hip_runtime.h>

// ---- CDNA5 types ----
typedef __attribute__((ext_vector_type(16))) _Float16     v16h;
typedef __attribute__((ext_vector_type(8)))  _Float16     v8h;
typedef __attribute__((ext_vector_type(8)))  float        v8f;
typedef __attribute__((ext_vector_type(4)))  unsigned int u32x4;
typedef __attribute__((ext_vector_type(8)))  int          i32x8;
typedef __attribute__((ext_vector_type(4)))  int          i32x4;

#define N_OUT  11008
#define K_IN   4096
#define M_ROWS 8192            // 4 * 2048
#define BM     128
#define BN     128
#define BK     32              // halves per K tile (one WMMA K step)
#define ROW_B  (BK * 2 + 16)   // 80 B: 64 B data + 16 B pad (alignment + bank spread)
#define TILE_B (BM * ROW_B)    // 10240 B
#define BUF_B  (2 * TILE_B)    // A tile + B tile per buffer
#define NKT    (K_IN / BK)     // 128 K tiles

union Frag16 {
  v16h v;
  v8h  h[2];
};

__device__ __forceinline__ v16h splat16(_Float16 s) {
  v16h r = {s, s, s, s, s, s, s, s, s, s, s, s, s, s, s, s};
  return r;
}

// Branch-free sign() on a packed f16 fragment: clamp(w * 2^12 * 2^12, -1, +1).
// Smallest f16 subnormal (2^-24) maps to exactly 1.0; overflow saturates via
// +-inf then min/max; zero stays zero. Lowers to v_pk_mul/min_num/max_num_f16.
__device__ __forceinline__ v16h sign16h(v16h w) {
  const _Float16 k = (_Float16)4096.0f;      // 2^12 (exact in f16)
  v16h t = w * k;
  t = t * k;                                  // keep separate: no reassociation
  t = __builtin_elementwise_min(t, splat16((_Float16)1.0f));
  t = __builtin_elementwise_max(t, splat16((_Float16)-1.0f));
  return t;
}

// TDM: DMA a 16-row x BK-half tile (row stride K_IN halves in memory) into LDS,
// padding 16 B after every 64 B row so LDS row stride is ROW_B.
__device__ __forceinline__ void tdm_load_16rows(const _Float16* gsrc, unsigned lds_off) {
  unsigned long long ga = (unsigned long long)(const void*)gsrc;
  u32x4 g0;
  g0[0] = 1u;                                              // count=1, user descriptor
  g0[1] = lds_off;                                         // lds_addr (bytes)
  g0[2] = (unsigned)ga;                                    // global_addr[31:0]
  g0[3] = ((unsigned)(ga >> 32) & 0x01FFFFFFu) | (2u << 30); // addr[56:32] | type=2
  i32x8 g1;
  g1[0] = (1 << 16)                                        // data_size = 2 B
        | (1 << 20)                                        // pad_enable
        | (3 << 22)                                        // pad_interval: 16 dwords (64 B)
        | (3 << 25);                                       // pad_amount: 4 dwords (16 B)
  g1[1] = (int)((K_IN & 0xFFFFu) << 16);                   // tensor_dim0 lo16
  g1[2] = (int)(((unsigned)K_IN >> 16) | (16u << 16));     // dim0 hi16 | tensor_dim1 lo16
  g1[3] = (int)(BK << 16);                                 // dim1 hi16 | tile_dim0 = 32
  g1[4] = 16;                                              // tile_dim1 = 16, tile_dim2 = 0
  g1[5] = K_IN;                                            // tensor_dim0_stride lo32
  g1[6] = 0;                                               // stride hi | dim1_stride lo
  g1[7] = 0;
  i32x4 gz4 = {0, 0, 0, 0};
  i32x8 gz8 = {0, 0, 0, 0, 0, 0, 0, 0};
  __builtin_amdgcn_tensor_load_to_lds(g0, g1, gz4, gz4, gz8, 0);
}

__global__ __launch_bounds__(256)
void BinaryLinear_26714696581575_kernel(const _Float16* __restrict__ x,
                                        const _Float16* __restrict__ w,
                                        const _Float16* __restrict__ bias,
                                        float* __restrict__ out) {
  __shared__ __align__(16) unsigned char smem[2 * BUF_B];  // 40 KB, double buffered

  const int tid  = threadIdx.x;
  const int wid  = __builtin_amdgcn_readfirstlane(tid >> 5); // wave 0..7 (uniform/SGPR)
  const int lane = tid & 31;
  const int lrow = lane & 15;
  const int sel  = lane >> 4;                                // lane half-group

  const int m0 = blockIdx.y * BM;
  const int n0 = blockIdx.x * BN;
  const int wm = (wid & 3) * 32;   // wave row offset inside tile
  const int wn = (wid >> 2) * 64;  // wave col offset inside tile

  const unsigned smem_lds = (unsigned)(unsigned long long)(void*)smem;

  // each wave DMAs its own 16-row slice of the A and B tiles
  const _Float16* xw   = x + (unsigned long long)(m0 + wid * 16) * K_IN;
  const _Float16* ww   = w + (unsigned long long)(n0 + wid * 16) * K_IN;
  const unsigned  aoff = smem_lds + (unsigned)(wid * 16 * ROW_B);
  const unsigned  boff = aoff + TILE_B;

  // prefetch K-tile 0 into buffer 0
  tdm_load_16rows(xw, aoff);
  tdm_load_16rows(ww, boff);

  v8f acc[2][4] = {};

  for (int kt = 0; kt < NKT; ++kt) {
    const int buf = kt & 1;
    if (kt + 1 < NKT) {
      const unsigned nxt = (unsigned)((buf ^ 1) * BUF_B);
      tdm_load_16rows(xw + (kt + 1) * BK, aoff + nxt);
      tdm_load_16rows(ww + (kt + 1) * BK, boff + nxt);
      __builtin_amdgcn_s_wait_tensorcnt(2);   // current pair done; next pair in flight
    } else {
      __builtin_amdgcn_s_wait_tensorcnt(0);
    }
    __syncthreads();                          // publish this wave's LDS slice to all

    const unsigned char* Ab = smem + buf * BUF_B;
    const unsigned char* Bb = Ab + TILE_B;

    // A fragments: lane holds row m = lane%16, 16 contiguous K at (lane<16 ? 0 : 16)
    Frag16 a[2];
#pragma unroll
    for (int mb = 0; mb < 2; ++mb) {
      const int off = (wm + mb * 16 + lrow) * ROW_B + sel * 32;
      a[mb].h[0] = *(const v8h*)(Ab + off);
      a[mb].h[1] = *(const v8h*)(Ab + off + 16);
    }
    // B fragments: lane holds col n = lane%16, 16 contiguous K at (lane<16 ? 0 : 16)
    Frag16 b[4];
#pragma unroll
    for (int nb = 0; nb < 4; ++nb) {
      const int off = (wn + nb * 16 + lrow) * ROW_B + sel * 32;
      b[nb].h[0] = *(const v8h*)(Bb + off);
      b[nb].h[1] = *(const v8h*)(Bb + off + 16);
      b[nb].v = sign16h(b[nb].v);             // fold sign(w), packed f16 ops
    }

#pragma unroll
    for (int mb = 0; mb < 2; ++mb)
#pragma unroll
      for (int nb = 0; nb < 4; ++nb)
        acc[mb][nb] = __builtin_amdgcn_wmma_f32_16x16x32_f16(
            false, a[mb].v, false, b[nb].v, (short)0, acc[mb][nb], false, false);

    __syncthreads();                          // buffer reuse fence for next DMA
  }

  // epilogue: C/D layout — VGPR j: (M = j + 8*sel, N = lane%16)
#pragma unroll
  for (int nb = 0; nb < 4; ++nb) {
    const int   col = n0 + wn + nb * 16 + lrow;
    const float bv  = (float)bias[col];
#pragma unroll
    for (int mb = 0; mb < 2; ++mb) {
      const int rbase = m0 + wm + mb * 16 + sel * 8;
#pragma unroll
      for (int j = 0; j < 8; ++j)
        out[(unsigned long long)(rbase + j) * N_OUT + col] = acc[mb][nb][j] + bv;
    }
  }
}

extern "C" void kernel_launch(void* const* d_in, const int* in_sizes, int n_in,
                              void* d_out, int out_size, void* d_ws, size_t ws_size,
                              hipStream_t stream) {
  (void)in_sizes; (void)n_in; (void)out_size; (void)d_ws; (void)ws_size;
  const _Float16* x    = (const _Float16*)d_in[0];
  const _Float16* w    = (const _Float16*)d_in[1];
  const _Float16* bias = (const _Float16*)d_in[2];
  float*          out  = (float*)d_out;
  dim3 grid(N_OUT / BN, M_ROWS / BM);   // 86 x 64 tiles
  BinaryLinear_26714696581575_kernel<<<grid, 256, 0, stream>>>(x, w, bias, out);
}
